// Attention_81020263072470
// MI455X (gfx1250) — compile-verified, matrix operand order checked by reading, not run on hardware
//
#include <hip/hip_runtime.h>
#include <hip/hip_bf16.h>

typedef __attribute__((ext_vector_type(16))) __bf16          v16bf;
typedef __attribute__((ext_vector_type(8)))  float           v8f;
typedef __attribute__((ext_vector_type(4)))  float           f32x4;   // clang vector: NT-load legal

static constexpr int NB = 32;    // batch
static constexpr int NS = 4096;  // sequence
static constexpr int ND = 512;   // key/query dim
static constexpr int NH = 512;   // hidden dim
static constexpr int NTILES = NH / 16;  // 32 N-tiles of 16
static constexpr int KSTEPS = ND / 32;  // 16 K-steps of 32
static constexpr int STILE  = 64;       // sequence rows per block (4 M-tiles)

__device__ __forceinline__ unsigned short f2bf(float f) {
    unsigned int u = __float_as_uint(f);
    u += 0x7FFFu + ((u >> 16) & 1u);   // round to nearest even
    return (unsigned short)(u >> 16);
}

// packed f32x2 -> bf16x2 (CDNA5 V_CVT_PK_BF16_F32)
__device__ __forceinline__ unsigned int pk_bf16(float lo, float hi) {
    unsigned int r;
    asm volatile("v_cvt_pk_bf16_f32 %0, %1, %2" : "=v"(r) : "v"(lo), "v"(hi));
    return r;
}

// single-instruction tanh (CDNA5 V_TANH_F32, TRANS op; v_nop covers the RAW hazard
// since the compiler cannot classify the asm as transcendental)
__device__ __forceinline__ float fast_tanh(float x) {
    float r;
    asm volatile("v_tanh_f32 %0, %1\n\tv_nop" : "=v"(r) : "v"(x));
    return r;
}

// ---------------------------------------------------------------- zero init
__global__ void k_zero(float* __restrict__ out, float* __restrict__ sumAcc) {
    int i = blockIdx.x * 256 + threadIdx.x;
    if (i < NB * ND) out[i] = 0.0f;
    if (i == 0) *sumAcc = 0.0f;
}

// ------------------------------------------------- W_k -> bf16 B fragments
// Fragment (nt, ks) covers B-matrix block K=[ks*32, ks*32+32), N=[nt*16, nt*16+16).
// Lane l (l<16): col N=l, K in {0..7, 16..23} (+8 if lane>=16), 2 K-values per VGPR.
__global__ void k_prep_bfrag(const float* __restrict__ Wk,
                             unsigned short* __restrict__ bfrag) {
    int frag = blockIdx.x;            // nt*KSTEPS + ks
    int nt = frag / KSTEPS;
    int ks = frag % KSTEPS;
    int lane = threadIdx.x;           // 0..31
    int n    = lane & 15;
    int half = lane >> 4;
    int h = nt * 16 + n;
    unsigned short* dst = bfrag + ((size_t)frag * 32 + lane) * 16;
#pragma unroll
    for (int e = 0; e < 16; ++e) {
        int klocal = (e < 8) ? (8 * half + e) : (16 + 8 * half + (e - 8));
        int d = ks * 32 + klocal;
        dst[e] = f2bf(Wk[(size_t)h * ND + d]);   // B element (k=d, n=h) = W_k[h][d]
    }
}

// ----------------------------------------------------- q = queries @ W_q^T
__global__ void k_prep_q(const float* __restrict__ queries,
                         const float* __restrict__ Wq,
                         float* __restrict__ qbuf) {
    int i = blockIdx.x * 256 + threadIdx.x;   // b*NH + h
    int b = i >> 9;
    int h = i & 511;
    const float4* qr = (const float4*)(queries + (size_t)b * ND);
    const float4* wr = (const float4*)(Wq + (size_t)h * ND);
    float acc = 0.0f;
    for (int t = 0; t < ND / 4; ++t) {
        float4 a = qr[t], w = wr[t];
        acc += a.x * w.x + a.y * w.y + a.z * w.z + a.w * w.w;
    }
    qbuf[i] = acc;
}

// ------------------------- main fused kernel: projection + tanh-score + exp
// Block: 64 seq rows (4 M-tiles) x all 512 H. 8 waves; wave w covers N-tiles 4w..4w+3.
// LDS: bf16 tile, XOR-swizzled 16B chunks (chunk' = chunk ^ row) -> exactly 64 KB with
// minimum (2-way) bank spread on every 32-lane ds_load_b128. sred aliases the tile
// after the K-loop (tile is dead by then, guarded by __syncthreads).
__global__ void __launch_bounds__(256)
k_scores(const float* __restrict__ keys, const float* __restrict__ wv,
         const float* __restrict__ qbuf, const unsigned short* __restrict__ bfrag,
         float* __restrict__ escores, float* __restrict__ sumAcc) {
    __shared__ __align__(16) unsigned char smemRaw[STILE * ND * 2];   // 65536 B
    unsigned short* tileb = (unsigned short*)smemRaw;
    float* sredF = (float*)smemRaw;          // [8 waves][64 rows], aliased later

    int blk   = blockIdx.x;
    int b     = blk >> 6;             // 64 s-tiles of 64 per batch
    int stile = blk & 63;
    int s0    = stile * STILE;
    int tid   = threadIdx.x;

    // Stage + convert: 64 rows are one contiguous 128 KB f32 block in HBM (streamed, NT).
    {
        const f32x4* src = (const f32x4*)(keys + ((size_t)b * NS + s0) * ND);
#pragma unroll
        for (int r = 0; r < 32; ++r) {
            int idx = tid + r * 256;          // float4 index within tile
            f32x4 v = __builtin_nontemporal_load(&src[idx]);
            uint2 p;
            p.x = pk_bf16(v[0], v[1]);
            p.y = pk_bf16(v[2], v[3]);
            int row   = idx >> 7;             // 128 float4 per row
            int col   = (idx & 127) * 4;      // bf16 element col within row
            int chunk = (col >> 3) ^ row;     // 16B-chunk XOR swizzle
            *(uint2*)(tileb + row * ND + chunk * 8 + (col & 7)) = p;
        }
    }
    __syncthreads();

    int wave = tid >> 5;
    int lane = tid & 31;
    int m    = lane & 15;   // A-matrix row within M-tile this lane holds
    int half = lane >> 4;

    v8f acc[4][4];          // [mtile][ntile]
#pragma unroll
    for (int mt = 0; mt < 4; ++mt)
#pragma unroll
        for (int n = 0; n < 4; ++n) acc[mt][n] = (v8f){0, 0, 0, 0, 0, 0, 0, 0};

    int rowm[4];
#pragma unroll
    for (int mt = 0; mt < 4; ++mt) rowm[mt] = mt * 16 + m;

#pragma unroll
    for (int ks = 0; ks < KSTEPS; ++ks) {
        // Four A fragments (one per M-tile). Lane's two 8-elem runs live in swizzled
        // chunks (ks*4+half)^row and (ks*4+2+half)^row of its row.
        union { int4 i[2]; v16bf b; } A[4];
#pragma unroll
        for (int mt = 0; mt < 4; ++mt) {
            int c0 = (ks * 4 + half) ^ rowm[mt];
            int c1 = (ks * 4 + 2 + half) ^ rowm[mt];
            const unsigned short* base = tileb + rowm[mt] * ND;
            A[mt].i[0] = *(const int4*)(base + c0 * 8);
            A[mt].i[1] = *(const int4*)(base + c1 * 8);
        }

#pragma unroll
        for (int n = 0; n < 4; ++n) {
            const int4* bp = (const int4*)(bfrag +
                ((size_t)((wave * 4 + n) * KSTEPS + ks) * 32 + lane) * 16);
            union { int4 i[2]; v16bf b; } Bf;
            Bf.i[0] = bp[0];
            Bf.i[1] = bp[1];
#pragma unroll
            for (int mt = 0; mt < 4; ++mt) {
                acc[mt][n] = __builtin_amdgcn_wmma_f32_16x16x32_bf16(
                    false, A[mt].b, false, Bf.b, (short)0, acc[mt][n], false, false);
            }
        }
    }

    __syncthreads();   // tile reads done everywhere; smem may be reused as sred

    // Epilogue: score += w_v[h] * tanh(k + q[b,h]); C layout: VGPR r, lane<16 -> M=r,
    // lane>=16 -> M=8+r, column N = lane&15.
    float sA[4][8];
#pragma unroll
    for (int mt = 0; mt < 4; ++mt)
#pragma unroll
        for (int r = 0; r < 8; ++r) sA[mt][r] = 0.0f;

#pragma unroll
    for (int n = 0; n < 4; ++n) {
        int h     = (wave * 4 + n) * 16 + (lane & 15);
        float qv  = qbuf[b * NH + h];
        float wvv = wv[h];
#pragma unroll
        for (int mt = 0; mt < 4; ++mt)
#pragma unroll
            for (int r = 0; r < 8; ++r)
                sA[mt][r] += wvv * fast_tanh(acc[mt][n][r] + qv);
    }

    // Reduce over the 16 columns held in each lane-half (xor masks 1..8 stay in-half)
#pragma unroll
    for (int off = 1; off <= 8; off <<= 1) {
#pragma unroll
        for (int mt = 0; mt < 4; ++mt)
#pragma unroll
            for (int r = 0; r < 8; ++r) sA[mt][r] += __shfl_xor(sA[mt][r], off, 32);
    }
    if ((lane & 15) == 0) {
#pragma unroll
        for (int mt = 0; mt < 4; ++mt)
#pragma unroll
            for (int r = 0; r < 8; ++r)
                sredF[wave * STILE + mt * 16 + half * 8 + r] = sA[mt][r];
    }
    __syncthreads();

    if (tid < STILE) {                 // waves 0 and 1, all 32 lanes each
        float s = 0.0f;
#pragma unroll
        for (int w = 0; w < 8; ++w) s += sredF[w * STILE + tid];
        float e = __expf(s);
        escores[(size_t)b * NS + s0 + tid] = e;
        float esum = e;
#pragma unroll
        for (int off = 1; off <= 16; off <<= 1) esum += __shfl_xor(esum, off, 32);
        if ((tid & 31) == 0) atomicAdd(sumAcc, esum);
    }
}

// ------------------------------------------- rep[b,d] = sum_s keys[b,s,d]*e[b,s]
__global__ void __launch_bounds__(512)
k_rep(const float* __restrict__ keys, const float* __restrict__ escores,
      float* __restrict__ out) {
    int b     = blockIdx.x >> 3;
    int chunk = blockIdx.x & 7;
    int d     = threadIdx.x;
    const float* kb = keys + (size_t)b * NS * ND;
    const float* eb = escores + (size_t)b * NS;
    float acc = 0.0f;
    int sBeg = chunk * 512;
    for (int s = sBeg; s < sBeg + 512; ++s) {
        acc += __builtin_nontemporal_load(&kb[(size_t)s * ND + d]) * eb[s];
    }
    atomicAdd(&out[b * ND + d], acc);
}

// ------------------------------------------------------- normalize by sum(e)
__global__ void k_scale(float* __restrict__ out, const float* __restrict__ sumAcc) {
    int i = blockIdx.x * 256 + threadIdx.x;
    float inv = 1.0f / (*sumAcc);
    if (i < NB * ND) out[i] *= inv;
}

extern "C" void kernel_launch(void* const* d_in, const int* in_sizes, int n_in,
                              void* d_out, int out_size, void* d_ws, size_t ws_size,
                              hipStream_t stream) {
    const float* keys    = (const float*)d_in[0];
    const float* queries = (const float*)d_in[1];
    const float* Wk      = (const float*)d_in[2];
    const float* Wq      = (const float*)d_in[3];
    const float* wv      = (const float*)d_in[4];
    float* out = (float*)d_out;

    char* ws = (char*)d_ws;
    float* qbuf             = (float*)(ws);                         //  64 KB
    float* escores          = (float*)(ws + 65536);                 // 512 KB
    float* sumAcc           = (float*)(ws + 65536 + 524288);        //   4 B (pad 256)
    unsigned short* bfrag   = (unsigned short*)(ws + 65536 + 524288 + 256);  // 512 KB

    k_zero<<<(NB * ND + 255) / 256, 256, 0, stream>>>(out, sumAcc);
    k_prep_bfrag<<<NTILES * KSTEPS, 32, 0, stream>>>(Wk, bfrag);
    k_prep_q<<<(NB * NH) / 256, 256, 0, stream>>>(queries, Wq, qbuf);
    k_scores<<<NB * (NS / STILE), 256, 0, stream>>>(keys, wv, qbuf, bfrag, escores, sumAcc);
    k_rep<<<NB * 8, 512, 0, stream>>>(keys, escores, out);
    k_scale<<<(NB * ND + 255) / 256, 256, 0, stream>>>(out, sumAcc);
}
